// PFLM_79817672229368
// MI455X (gfx1250) — matrix operand order repaired and checked by reading, not compiled, not run on hardware
//
// FIVO particle-filter LSTM LM for MI455X (gfx1250), bf16 WMMA everywhere.
//
// Roofline: 276 GFLOP total, 73% in the [1024,512]x[512,8000] output
// projection. bf16 weights (11.3 MB) stay L2-resident (192 MB L2); HBM
// traffic is a one-time ~33 MB -> the workload is compute/latency bound,
// so every GEMM uses v_wmma_f32_16x16x32_bf16 (f32 accumulate), and the
// log-softmax NLL is fused into the big GEMM as a running logsumexp
// (never materializes the 32 MB logits tensor).
//
// v2: activations are written in dual precision (f32 state + bf16 GEMM
// operand) by their producer kernels, so GEMM inner loops are pure
// load+wmma (no v_cvt in the hot path, no cross-block redundant cvt).
// The logits kernel is K-outer with 5 accumulators: one A-fragment load
// feeds 5 WMMAs. W_out row streams get global_prefetch_b8 warm-ups.
//
// Workspace requirement: ~41 MB in d_ws.

#include <hip/hip_runtime.h>
#include <hip/hip_bf16.h>
#include <math.h>

typedef __attribute__((ext_vector_type(16))) __bf16 v16bf;
typedef __attribute__((ext_vector_type(8)))  __bf16 v8bf;
typedef __attribute__((ext_vector_type(8)))  float  v8f;

#define LOG2PI_F 1.8378770664093453f

// ---------------- deterministic RNG (hash -> uniform / Box-Muller) --------
__device__ __forceinline__ unsigned hash_u32(unsigned x) {
  x ^= x >> 16; x *= 0x7feb352dU; x ^= x >> 15; x *= 0x846ca68bU; x ^= x >> 16;
  return x;
}
__device__ __forceinline__ float uniform01(unsigned s) {
  return ((hash_u32(s) >> 8) + 0.5f) * (1.0f / 16777216.0f);
}
__device__ __forceinline__ float normal01(unsigned s) {
  float u1 = uniform01(s ^ 0x68bc21ebU);
  float u2 = uniform01(s + 0x9e3779b9U);
  return sqrtf(fmaxf(-2.0f * __logf(u1), 0.0f)) * __cosf(6.28318530718f * u2);
}

// A-fragment load: lane(m) holds K = hf*8+0..7 and 16+hf*8+0..7 (ISA layout)
__device__ __forceinline__ v16bf load_a_frag(const __bf16* Arow, int k0, int hf) {
  const v8bf alo = *(const v8bf*)(Arow + k0 + hf * 8);
  const v8bf ahi = *(const v8bf*)(Arow + k0 + 16 + hf * 8);
  v16bf a;
#pragma unroll
  for (int e = 0; e < 8; ++e) { a[e] = alo[e]; a[e + 8] = ahi[e]; }
  return a;
}

// ---------------- generic bf16 WMMA GEMM:  C = A[bf16] @ W[bf16]^T ---------
// M fixed = 1024. Block = 4 waves, each wave owns one 16x16 output tile.
// Epilogue adds bias0[n]+bias1[n] and a rank-1 term (row%32)*rank1[n]
// (folds in the reference's constant-batch-index input matmuls).
__global__ __launch_bounds__(128) void gemm_bf16_wmma(
    const __bf16* __restrict__ A, const __bf16* __restrict__ W,
    float* __restrict__ C, const float* __restrict__ bias0,
    const float* __restrict__ bias1, const float* __restrict__ rank1,
    int N, int K, int accum)
{
  const int wave = threadIdx.x >> 5;
  const int lane = threadIdx.x & 31;
  const int hf   = lane >> 4;   // which 16-lane half
  const int ln   = lane & 15;
  const int mtile = blockIdx.x * 16;
  const int ntile = (blockIdx.y * 4 + wave) * 16;

  const __bf16* Arow = A + (size_t)(mtile + ln) * K;
  const __bf16* Wrow = W + (size_t)(ntile + ln) * K;

  v8f acc = {};
  for (int k0 = 0; k0 < K; k0 += 32) {
    const v16bf avec = load_a_frag(Arow, k0, hf);
    // B 32x16 frag: lane(n=ln) holds K = hf*8+0..7 and 16+hf*8+0..7 (same as A)
    const v16bf bvec = load_a_frag(Wrow, k0, hf);
    acc = __builtin_amdgcn_wmma_f32_16x16x32_bf16(false, avec, false, bvec,
                                                  (short)0, acc, false, false);
  }

  const int n = ntile + ln;
  float badd = 0.f;
  if (bias0) badd += bias0[n];
  if (bias1) badd += bias1[n];
  const float r1 = rank1 ? rank1[n] : 0.f;
#pragma unroll
  for (int r = 0; r < 8; ++r) {      // D layout: M = r + 8*hf, N = ln
    const int m = mtile + r + hf * 8;
    const size_t o = (size_t)m * N + n;
    if (accum) C[o] += acc[r];
    else       C[o]  = acc[r] + badd + (float)(m & 31) * r1;
  }
}

// ------- fused output-projection + running logsumexp partials -------------
// grid = (64 row-tiles, 25 V-chunks); each block: 4 waves x 5 tiles of 16
// vocab columns, K=512.  K-outer loop: one A fragment feeds 5 WMMAs into 5
// accumulators.  Then per-lane online (max,sumexp), butterfly-merge over the
// 16 lanes of each half, LDS-merge over 4 waves -> per-(rowtile,chunk) pair.
__global__ __launch_bounds__(128) void logits_lse_partial(
    const __bf16* __restrict__ Dh, const __bf16* __restrict__ Wout,
    const float* __restrict__ bout, float* __restrict__ part)
{
  __shared__ float smx[4][16];
  __shared__ float ssm[4][16];
  const int wave = threadIdx.x >> 5;
  const int lane = threadIdx.x & 31;
  const int hf = lane >> 4;
  const int ln = lane & 15;
  const int mtile = blockIdx.x * 16;
  const __bf16* Arow = Dh + (size_t)(mtile + ln) * 512;

  int ntile[5];
  const __bf16* Wrow[5];
#pragma unroll
  for (int tt = 0; tt < 5; ++tt) {
    ntile[tt] = (blockIdx.y * 20 + wave * 5 + tt) * 16;
    Wrow[tt] = Wout + (size_t)(ntile[tt] + ln) * 512;
    __builtin_prefetch(Wrow[tt], 0, 0);        // global_prefetch_b8 warm-up
  }

  v8f acc[5] = {{}, {}, {}, {}, {}};
  for (int k0 = 0; k0 < 512; k0 += 32) {
    const v16bf avec = load_a_frag(Arow, k0, hf);
#pragma unroll
    for (int tt = 0; tt < 5; ++tt) {
      const v16bf bvec = load_a_frag(Wrow[tt], k0, hf);
      acc[tt] = __builtin_amdgcn_wmma_f32_16x16x32_bf16(false, avec, false, bvec,
                                                        (short)0, acc[tt], false, false);
    }
  }

  float mx[8], sm[8];
#pragma unroll
  for (int r = 0; r < 8; ++r) { mx[r] = -3.4e38f; sm[r] = 0.f; }
#pragma unroll
  for (int tt = 0; tt < 5; ++tt) {
    const float cb = bout[ntile[tt] + ln];
#pragma unroll
    for (int r = 0; r < 8; ++r) {
      const float v = acc[tt][r] + cb;
      if (v > mx[r]) { sm[r] = sm[r] * __expf(mx[r] - v) + 1.f; mx[r] = v; }
      else           sm[r] += __expf(v - mx[r]);
    }
  }
  // merge across the 16 lanes that share each row (xor stays within a half)
  for (int msk = 1; msk < 16; msk <<= 1) {
#pragma unroll
    for (int r = 0; r < 8; ++r) {
      const float om = __shfl_xor(mx[r], msk, 32);
      const float os = __shfl_xor(sm[r], msk, 32);
      if (om > mx[r]) { sm[r] = sm[r] * __expf(mx[r] - om) + os; mx[r] = om; }
      else            sm[r] += os * __expf(om - mx[r]);
    }
  }
  if (ln == 0) {
#pragma unroll
    for (int r = 0; r < 8; ++r) { smx[wave][hf * 8 + r] = mx[r]; ssm[wave][hf * 8 + r] = sm[r]; }
  }
  __syncthreads();
  if (threadIdx.x < 16) {
    const int row = threadIdx.x;
    float M = smx[0][row], S = ssm[0][row];
    for (int w2 = 1; w2 < 4; ++w2) {
      const float om = smx[w2][row], os = ssm[w2][row];
      if (om > M) { S = S * __expf(M - om) + os; M = om; }
      else        S += os * __expf(om - M);
    }
    float* p = part + (((size_t)blockIdx.x * gridDim.y + blockIdx.y) * 16 + row) * 2;
    p[0] = M; p[1] = S;
  }
}

// merge 25 partials/row -> lse, nll = lse - logit_tgt; accumulate nll_total
__global__ __launch_bounds__(256) void lse_finalize(
    const float* __restrict__ part, const float* __restrict__ ltgt,
    float* __restrict__ nll, float* __restrict__ dout, int nvb)
{
  __shared__ float sred[256];
  const int row = blockIdx.x * 256 + threadIdx.x;
  const int rt = row >> 4, rr = row & 15;
  float M = -3.4e38f, S = 0.f;
  for (int vb = 0; vb < nvb; ++vb) {
    const float* p = part + (((size_t)rt * nvb + vb) * 16 + rr) * 2;
    const float m = p[0], s = p[1];
    if (m > M) { S = S * __expf(M - m) + s; M = m; }
    else       S += s * __expf(m - M);
  }
  const float v = M + __logf(S) - ltgt[row];
  nll[row] = v;
  sred[threadIdx.x] = v; __syncthreads();
  for (int s2 = 128; s2 > 0; s2 >>= 1) {
    if (threadIdx.x < s2) sred[threadIdx.x] += sred[threadIdx.x + s2];
    __syncthreads();
  }
  if (threadIdx.x == 0) atomicAdd(dout + 1, sred[0] * (1.0f / 32.0f)); // mean over B
}

// exact f32 target logit: one wave per row, dot over K=512
__global__ __launch_bounds__(128) void logit_tgt_kernel(
    const float* __restrict__ dh, const float* __restrict__ Wout,
    const float* __restrict__ bout, const int* __restrict__ tokens,
    float* __restrict__ out, int t)
{
  const int wave = threadIdx.x >> 5, lane = threadIdx.x & 31;
  const int row = blockIdx.x * 4 + wave;
  const int tgt = tokens[t * 32 + (row & 31)];
  const float* a = dh + (size_t)row * 512;
  const float* w = Wout + (size_t)tgt * 512;
  float s = 0.f;
  for (int k = lane; k < 512; k += 32) s += a[k] * w[k];
  for (int m = 16; m > 0; m >>= 1) s += __shfl_xor(s, m, 32);
  if (lane == 0) out[row] = s + bout[tgt];
}

// ---------------- elementwise / small kernels -----------------------------
// dual-precision h output: f32 for state math, bf16 for WMMA consumers
__global__ __launch_bounds__(256) void lstm_cell_kernel(
    const float* __restrict__ g, const float* __restrict__ cin,
    float* __restrict__ hout, float* __restrict__ cout,
    __bf16* __restrict__ houtbf, int H)
{
  const int i = blockIdx.x * 256 + threadIdx.x;       // grid = BP*H exactly
  const int row = i / H, j = i - row * H;
  const float* gr = g + (size_t)row * 4 * H;
  const float ig = 1.f / (1.f + __expf(-gr[j]));
  const float fg = 1.f / (1.f + __expf(-gr[H + j]));
  const float gg = tanhf(gr[2 * H + j]);
  const float og = 1.f / (1.f + __expf(-gr[3 * H + j]));
  const float cn = fg * cin[i] + ig * gg;
  cout[i] = cn;
  const float hv = og * tanhf(cn);
  hout[i] = hv;
  if (houtbf) houtbf[i] = (__bf16)hv;
}

__global__ __launch_bounds__(128) void sample_kernel(
    const float* __restrict__ mean, const float* __restrict__ logvar,
    float* __restrict__ z, __bf16* __restrict__ zbf,
    float* __restrict__ slv, float* __restrict__ se2, int t)
{
  __shared__ float sa[128], sb[128];
  const int row = blockIdx.x, k = threadIdx.x;
  const size_t i = (size_t)row * 128 + k;
  const float lv = logvar[i];
  const unsigned seed = (unsigned)(t * 16777619) ^ (unsigned)(i * 2654435761u) ^ 0xA511E9B3u;
  const float eps = normal01(seed);
  const float zv = eps * __expf(0.5f * lv) + mean[i];
  z[i] = zv;
  zbf[i] = (__bf16)zv;
  sa[k] = lv; sb[k] = eps * eps; __syncthreads();
  for (int s = 64; s > 0; s >>= 1) {
    if (k < s) { sa[k] += sa[k + s]; sb[k] += sb[k + s]; }
    __syncthreads();
  }
  if (k == 0) { slv[row] = sa[0]; se2[row] = sb[0]; }
}

__global__ __launch_bounds__(128) void w_kernel(
    const float* __restrict__ nll, const float* __restrict__ z,
    const float* __restrict__ ph, const float* __restrict__ slv,
    const float* __restrict__ se2, float* __restrict__ wout)
{
  __shared__ float sb[128];
  const int row = blockIdx.x, k = threadIdx.x;
  const float d = z[(size_t)row * 128 + k] - ph[(size_t)row * 128 + k];
  sb[k] = d * d; __syncthreads();
  for (int s = 64; s > 0; s >>= 1) { if (k < s) sb[k] += sb[k + s]; __syncthreads(); }
  if (k == 0) {
    const float f = -0.5f * LOG2PI_F - 0.5f * sb[0];
    const float r = -0.5f * 128.0f * LOG2PI_F - slv[row] - 0.5f * se2[row];
    wout[row] = -nll[row] + f - r;
  }
}

__global__ __launch_bounds__(32) void fivo_kernel(const float* __restrict__ w,
                                                  float* __restrict__ dout)
{
  const int b = threadIdx.x;
  float mx = -3.4e38f;
  for (int p = 0; p < 32; ++p) mx = fmaxf(mx, w[p * 32 + b]);
  float s = 0.f;
  for (int p = 0; p < 32; ++p) s += __expf(w[p * 32 + b] - mx);
  float c = mx + __logf(s) - 3.4657359028f;              // - log(P=32)
  for (int m = 16; m > 0; m >>= 1) c += __shfl_xor(c, m, 32);
  if (b == 0) atomicAdd(dout, -c);
}

__global__ __launch_bounds__(32) void resample_kernel(const float* __restrict__ w,
                                                      int* __restrict__ idx, int t)
{
  __shared__ float probs[32];
  const int b = blockIdx.x, p = threadIdx.x;
  const float wv = w[p * 32 + b];
  float mx = wv;
  for (int m = 16; m > 0; m >>= 1) mx = fmaxf(mx, __shfl_xor(mx, m, 32));
  const float e = __expf(wv - mx);
  float s = e;
  for (int m = 16; m > 0; m >>= 1) s += __shfl_xor(s, m, 32);
  float pr = e / s + 0.01f;
  float tot = pr;
  for (int m = 16; m > 0; m >>= 1) tot += __shfl_xor(tot, m, 32);
  probs[p] = pr / tot;
  __syncthreads();
  const unsigned seed = 0x51A7C0DEu ^ (unsigned)(t * 92837111) ^
                        (unsigned)(p * 689287499) ^ (unsigned)(b * 283923481);
  const float u = uniform01(seed);
  float c = 0.f; int anc = 31;
  for (int q = 0; q < 32; ++q) { c += probs[q]; if (u < c) { anc = q; break; } }
  idx[p * 32 + b] = anc * 32 + b;
}

// optional bf16 dual output for gathered states consumed by WMMA GEMMs
__global__ __launch_bounds__(256) void gather_kernel(const float* __restrict__ src,
    float* __restrict__ dst, __bf16* __restrict__ dstbf,
    const int* __restrict__ idx, int width)
{
  const int i = blockIdx.x * 256 + threadIdx.x;        // grid = BP*width exactly
  const int row = i / width;
  const int k = i - row * width;
  const float v = src[(size_t)idx[row] * width + k];
  dst[i] = v;
  if (dstbf) dstbf[i] = (__bf16)v;
}

// ---------------- setup kernels -------------------------------------------
__global__ __launch_bounds__(256) void cvt_bf16_kernel(const float* __restrict__ src,
    __bf16* __restrict__ dst, long total, int cols, int ld)
{
  const long i = (long)blockIdx.x * 256 + threadIdx.x;
  if (i >= total) return;
  const long r = i / cols, c = i - r * cols;
  dst[i] = (__bf16)src[r * (long)ld + c];
}

__global__ __launch_bounds__(256) void rowsum_kernel(const float* __restrict__ src,
    float* __restrict__ dst, int rows, int ld, int c0, int nc)
{
  const int r = blockIdx.x * 256 + threadIdx.x;
  if (r >= rows) return;
  const float* p = src + (size_t)r * ld + c0;
  float s = 0.f;
  for (int c = 0; c < nc; ++c) s += p[c];
  dst[r] = s;
}

__global__ __launch_bounds__(256) void zero_kernel(float* __restrict__ p, long n)
{
  const long i = (long)blockIdx.x * 256 + threadIdx.x;
  if (i < n) p[i] = 0.f;
}

// ---------------- host driver ---------------------------------------------
extern "C" void kernel_launch(void* const* d_in, const int* in_sizes, int n_in,
                              void* d_out, int out_size, void* d_ws, size_t ws_size,
                              hipStream_t stream) {
  const int S = 24, B = 32, BP = 1024, Z = 128, NH = 512;
  (void)in_sizes; (void)n_in; (void)out_size; (void)ws_size;

  const int*   tokens   = (const int*)d_in[0];
  const float* W_ih_z   = (const float*)d_in[1];
  const float* W_hh_z   = (const float*)d_in[2];
  const float* b_ih_z   = (const float*)d_in[3];
  const float* b_hh_z   = (const float*)d_in[4];
  const float* W_mean   = (const float*)d_in[5];
  const float* b_mean   = (const float*)d_in[6];
  const float* W_logvar = (const float*)d_in[7];
  const float* b_logvar = (const float*)d_in[8];
  const float* W_ih_p   = (const float*)d_in[9];
  const float* W_hh_p   = (const float*)d_in[10];
  const float* b_ih_p   = (const float*)d_in[11];
  const float* b_hh_p   = (const float*)d_in[12];
  const float* W_ih_d   = (const float*)d_in[13];
  const float* W_hh_d   = (const float*)d_in[14];
  const float* b_ih_d   = (const float*)d_in[15];
  const float* b_hh_d   = (const float*)d_in[16];
  const float* W_out    = (const float*)d_in[17];
  const float* b_out    = (const float*)d_in[18];

  // ---- workspace carve-up (256B aligned) ----
  char* wsb = (char*)d_ws;
  size_t off = 0;
  auto alloc = [&](size_t bytes) -> char* {
    char* p = wsb + off;
    off = (off + bytes + 255) & ~(size_t)255;
    return p;
  };
  __bf16* bWz   = (__bf16*)alloc((size_t)2 * 512 * 128);
  __bf16* bWm   = (__bf16*)alloc((size_t)2 * 128 * 128);
  __bf16* bWl   = (__bf16*)alloc((size_t)2 * 128 * 128);
  __bf16* bWpi  = (__bf16*)alloc((size_t)2 * 512 * 128);
  __bf16* bWph  = (__bf16*)alloc((size_t)2 * 512 * 128);
  __bf16* bWdz  = (__bf16*)alloc((size_t)2 * 2048 * 128);
  __bf16* bWdh  = (__bf16*)alloc((size_t)2 * 2048 * 512);
  __bf16* bWout = (__bf16*)alloc((size_t)2 * 8000 * 512);
  float* rs_z   = (float*)alloc(4 * 512);
  float* rs_d   = (float*)alloc(4 * 2048);
  const long STATEF = 4L * BP * Z + 2L * BP * NH;   // h,c,ph,pc,dh,dc
  float* stateA = (float*)alloc(4 * STATEF);
  float* stateB = (float*)alloc(4 * STATEF);
  float* henc   = (float*)alloc(4L * BP * Z);
  float* meanb  = (float*)alloc(4L * BP * Z);
  float* logvb  = (float*)alloc(4L * BP * Z);
  float* zbuf   = (float*)alloc(4L * BP * Z);
  float* g_enc  = (float*)alloc(4L * BP * 512);
  float* g_dec  = (float*)alloc(4L * BP * 2048);
  float* g_p    = (float*)alloc(4L * BP * 512);
  // bf16 activation mirrors (producers write them; GEMMs consume them)
  __bf16* bhg      = (__bf16*)alloc((size_t)2 * BP * Z);   // gathered z (enc & prior input)
  __bf16* bhenc    = (__bf16*)alloc((size_t)2 * BP * Z);   // encoder h
  __bf16* bz       = (__bf16*)alloc((size_t)2 * BP * Z);   // sampled z
  __bf16* bphg     = (__bf16*)alloc((size_t)2 * BP * Z);   // gathered p_h
  __bf16* bdh      = (__bf16*)alloc((size_t)2 * BP * NH);  // carry d_h (pre-cell)
  __bf16* bdh_post = (__bf16*)alloc((size_t)2 * BP * NH);  // d_h after cell (logits input)
  float* aux_slv = (float*)alloc(4 * BP);
  float* aux_se2 = (float*)alloc(4 * BP);
  float* ltgt    = (float*)alloc(4 * BP);
  float* nllb    = (float*)alloc(4 * BP);
  float* wbuf    = (float*)alloc(4 * BP);
  float* lsep    = (float*)alloc(4L * 64 * 25 * 16 * 2);
  int*   idxb    = (int*)alloc(4 * BP);

  float* setA[6], *setB[6];
  setA[0] = stateA;                 setB[0] = stateB;                 // h
  setA[1] = stateA + 1L * BP * Z;   setB[1] = stateB + 1L * BP * Z;   // c
  setA[2] = stateA + 2L * BP * Z;   setB[2] = stateB + 2L * BP * Z;   // p_h
  setA[3] = stateA + 3L * BP * Z;   setB[3] = stateB + 3L * BP * Z;   // p_c
  setA[4] = stateA + 4L * BP * Z;   setB[4] = stateB + 4L * BP * Z;   // d_h
  setA[5] = setA[4] + 1L * BP * NH; setB[5] = setB[4] + 1L * BP * NH; // d_c

  // ---- setup (re-done every call for determinism; cheap) ----
  zero_kernel<<<1, 256, 0, stream>>>((float*)d_out, 2);
  zero_kernel<<<(unsigned)((STATEF + 255) / 256), 256, 0, stream>>>(stateA, STATEF);
  zero_kernel<<<(BP * Z / 2) / 256, 256, 0, stream>>>((float*)bhg, BP * Z / 2);   // bf16 zeros
  zero_kernel<<<(BP * NH / 2) / 256, 256, 0, stream>>>((float*)bdh, BP * NH / 2); // bf16 zeros
  auto cvt = [&](const float* s, __bf16* d, long rows, int cols, int ld) {
    long tot = rows * cols;
    cvt_bf16_kernel<<<(unsigned)((tot + 255) / 256), 256, 0, stream>>>(s, d, tot, cols, ld);
  };
  cvt(W_hh_z,   bWz,   512, 128, 128);
  cvt(W_mean,   bWm,   128, 128, 128);
  cvt(W_logvar, bWl,   128, 128, 128);
  cvt(W_ih_p,   bWpi,  512, 128, 128);
  cvt(W_hh_p,   bWph,  512, 128, 128);
  cvt(W_ih_d,   bWdz,  2048, 128, 384);   // z-columns slice of W_ih_d
  cvt(W_hh_d,   bWdh,  2048, 512, 512);
  cvt(W_out,    bWout, 8000, 512, 512);
  rowsum_kernel<<<2, 256, 0, stream>>>(W_ih_z, rs_z, 512, 1024, 0, 1024);
  rowsum_kernel<<<8, 256, 0, stream>>>(W_ih_d, rs_d, 2048, 384, 128, 256);

  auto gemm = [&](const __bf16* A, const __bf16* W, float* C, const float* b0,
                  const float* b1, const float* r1, int N, int K, int accum) {
    gemm_bf16_wmma<<<dim3(BP / 16, N / 64), 128, 0, stream>>>(A, W, C, b0, b1, r1, N, K, accum);
  };

  float** cur = setA;
  float** nxt = setB;
  for (int t = 0; t < S; ++t) {
    float *c = cur[1], *ph = cur[2], *pc = cur[3], *dh = cur[4], *dc = cur[5];
    // encoder LSTM: rank-1 input term + recurrent WMMA GEMM (A = bhg = bf16 h)
    gemm(bhg, bWz, g_enc, b_ih_z, b_hh_z, rs_z, 512, 128, 0);
    lstm_cell_kernel<<<(BP * Z) / 256, 256, 0, stream>>>(g_enc, c, henc, c, bhenc, Z);
    // mean / logvar
    gemm(bhenc, bWm, meanb, b_mean,   nullptr, nullptr, 128, 128, 0);
    gemm(bhenc, bWl, logvb, b_logvar, nullptr, nullptr, 128, 128, 0);
    sample_kernel<<<BP, 128, 0, stream>>>(meanb, logvb, zbuf, bz, aux_slv, aux_se2, t);
    // decoder LSTM
    gemm(bz,  bWdz, g_dec, b_ih_d, b_hh_d, rs_d, 2048, 128, 0);
    gemm(bdh, bWdh, g_dec, nullptr, nullptr, nullptr, 2048, 512, 1);
    lstm_cell_kernel<<<(BP * NH) / 256, 256, 0, stream>>>(g_dec, dc, dh, dc, bdh_post, NH);
    // fused output projection -> nll
    logit_tgt_kernel<<<BP / 4, 128, 0, stream>>>(dh, W_out, b_out, tokens, ltgt, t);
    logits_lse_partial<<<dim3(64, 25), 128, 0, stream>>>(bdh_post, bWout, b_out, lsep);
    lse_finalize<<<4, 256, 0, stream>>>(lsep, ltgt, nllb, (float*)d_out, 25);
    // importance weights, fivo, resample, gather
    w_kernel<<<BP, 128, 0, stream>>>(nllb, zbuf, ph, aux_slv, aux_se2, wbuf);
    fivo_kernel<<<1, 32, 0, stream>>>(wbuf, (float*)d_out);
    resample_kernel<<<B, 32, 0, stream>>>(wbuf, idxb, t);
    gather_kernel<<<(BP * Z)  / 256, 256, 0, stream>>>(zbuf, nxt[0], bhg,  idxb, Z);
    gather_kernel<<<(BP * Z)  / 256, 256, 0, stream>>>(c,    nxt[1], (__bf16*)nullptr, idxb, Z);
    gather_kernel<<<(BP * Z)  / 256, 256, 0, stream>>>(ph,   nxt[2], bphg, idxb, Z);
    gather_kernel<<<(BP * Z)  / 256, 256, 0, stream>>>(pc,   nxt[3], (__bf16*)nullptr, idxb, Z);
    gather_kernel<<<(BP * NH) / 256, 256, 0, stream>>>(dh,   nxt[4], bdh,  idxb, NH);
    gather_kernel<<<(BP * NH) / 256, 256, 0, stream>>>(dc,   nxt[5], (__bf16*)nullptr, idxb, NH);
    // autoregressive prior LSTM on gathered states
    gemm(bhg,  bWpi, g_p, b_ih_p, b_hh_p, nullptr, 512, 128, 0);
    gemm(bphg, bWph, g_p, nullptr, nullptr, nullptr, 512, 128, 1);
    lstm_cell_kernel<<<(BP * Z) / 256, 256, 0, stream>>>(g_p, nxt[3], nxt[2], nxt[3],
                                                         (__bf16*)nullptr, Z);
    // swap state sets
    float** tmp = cur; cur = nxt; nxt = tmp;
  }
}
